// PointConvNet_33105607917645
// MI455X (gfx1250) — matrix operand order repaired and checked by reading, not compiled, not run on hardware
//
#include <hip/hip_runtime.h>

typedef __attribute__((ext_vector_type(16))) _Float16 v16h;
typedef __attribute__((ext_vector_type(8)))  _Float16 v8h;
typedef __attribute__((ext_vector_type(8)))  float    v8f;
typedef __attribute__((ext_vector_type(4)))  float    v4f;

#define NN    50000
#define NE    1600000
#define FIN   16
#define HID   64
#define FOUT  64

// float atomic max via int/uint monotonic bit trick (valid with -inf init):
// positive floats: int compare order == float order  -> atomicMax(int)
// negative floats: uint compare order is reversed    -> atomicMin(uint)
__device__ __forceinline__ void atomic_max_f32(float* addr, float val) {
  if (val >= 0.0f) {
    atomicMax((int*)addr, __float_as_int(val));
  } else {
    atomicMin((unsigned int*)addr, (unsigned int)__float_as_int(val));
  }
}

__global__ void __launch_bounds__(256) init_out_kernel(float* out, int n) {
  int i = blockIdx.x * blockDim.x + threadIdx.x;
  int stride = gridDim.x * blockDim.x;
  const float ninf = __int_as_float(0xFF800000);
  for (; i < n; i += stride) out[i] = ninf;
}

__global__ void __launch_bounds__(256) finalize_kernel(float* out, int n) {
  int i = blockIdx.x * blockDim.x + threadIdx.x;
  int stride = gridDim.x * blockDim.x;
  for (; i < n; i += stride) {
    float v = out[i];
    if (__float_as_uint(v) == 0xFF800000u) out[i] = 0.0f;  // no-edge node -> 0
  }
}

__global__ void __launch_bounds__(256)
pointconv_wmma_kernel(const float* __restrict__ x,
                      const float* __restrict__ pos,
                      const int*   __restrict__ eidx,   // [2, E]: src row then dst row
                      const float* __restrict__ W1,     // [19, 64]
                      const float* __restrict__ b1,     // [64]
                      const float* __restrict__ W2,     // [64, 64]
                      const float* __restrict__ b2,     // [64]
                      float* __restrict__ out)          // [N, 64]
{
  __shared__ __align__(16) _Float16 lds_h[8][16 * HID]; // 2KB per wave, 8 waves

  const int lane  = threadIdx.x & 31;
  const int wid   = threadIdx.x >> 5;
  const bool hi   = lane >= 16;
  const int col   = lane & 15;        // N column (C layout) / M row (A layout)
  const int mbase = hi ? 8 : 0;

  // ---- Preload W1 as 4 B-tiles (B layout: lane holds column N=nb*16+col, K elems {mbase+0..7, 16+mbase+0..7}) ----
  // Rows K >= FIN+3 are zeroed.
  v16h bw1[4];
  {
    #pragma unroll
    for (int nb = 0; nb < 4; ++nb) {
      v16h t;
      #pragma unroll
      for (int i = 0; i < 16; ++i) {
        const int k = (i < 8) ? (mbase + i) : (16 + mbase + i - 8);
        t[i] = (k < FIN + 3) ? (_Float16)W1[k * HID + nb * 16 + col] : (_Float16)0.0f;
      }
      bw1[nb] = t;
    }
  }
  // ---- Preload W2 as 2 K-chunks x 4 N-tiles (all rows valid) ----
  v16h bw2[2][4];
  #pragma unroll
  for (int kc = 0; kc < 2; ++kc) {
    #pragma unroll
    for (int nb = 0; nb < 4; ++nb) {
      v16h t;
      #pragma unroll
      for (int i = 0; i < 16; ++i) {
        const int k = kc * 32 + ((i < 8) ? (mbase + i) : (16 + mbase + i - 8));
        t[i] = (_Float16)W2[k * HID + nb * 16 + col];
      }
      bw2[kc][nb] = t;
    }
  }
  // ---- Biases: column-dependent only -> one scalar per lane per N-block ----
  float b1v[4], b2v[4];
  #pragma unroll
  for (int nb = 0; nb < 4; ++nb) {
    b1v[nb] = b1[nb * 16 + col];
    b2v[nb] = b2[nb * 16 + col];
  }

  _Float16* hbuf = lds_h[wid];
  const int ntiles  = (NE + 15) / 16;
  const int tstride = gridDim.x * 8;

  for (int tile = blockIdx.x * 8 + wid; tile < ntiles; tile += tstride) {
    // ================= Build A1 = [x_src | pos_src - pos_dst | 0pad] =================
    // A layout (16-bit, 16x32): lane<16 -> row=lane,   K elems {0..7, 16..23}
    //                           lane>=16 -> row=lane-16, K elems {8..15, 24..31}
    // Branchless: both half-waves issue the same-shape vector loads.
    int erow   = tile * 16 + col;
    int eclamp = erow < NE ? erow : NE - 1;
    int s = eidx[eclamp];
    int d = eidx[NE + eclamp];

    const int xoff = s * FIN + (hi ? 8 : 0);
    v4f x0 = *(const v4f*)(x + xoff);
    v4f x1 = *(const v4f*)(x + xoff + 4);
    // all lanes load pos (high lanes hit the same cachelines as their low twin)
    float r0 = pos[s * 3 + 0] - pos[d * 3 + 0];
    float r1 = pos[s * 3 + 1] - pos[d * 3 + 1];
    float r2 = pos[s * 3 + 2] - pos[d * 3 + 2];

    v16h a1;
    #pragma unroll
    for (int i = 0; i < 4; ++i) {
      a1[i]     = (_Float16)x0[i];
      a1[4 + i] = (_Float16)x1[i];
    }
    a1[8]  = (_Float16)(hi ? 0.0f : r0);   // K=16 (rel.x) on low half, K=24 pad on high
    a1[9]  = (_Float16)(hi ? 0.0f : r1);
    a1[10] = (_Float16)(hi ? 0.0f : r2);
    #pragma unroll
    for (int i = 11; i < 16; ++i) a1[i] = (_Float16)0.0f;

    // ================= GEMM1 (19->64) + bias + ReLU, stage h in LDS =================
    #pragma unroll
    for (int nb = 0; nb < 4; ++nb) {
      v8f c;
      #pragma unroll
      for (int v = 0; v < 8; ++v) c[v] = b1v[nb];
      c = __builtin_amdgcn_wmma_f32_16x16x32_f16(false, a1, false, bw1[nb],
                                                 (short)0, c, false, false);
      #pragma unroll
      for (int v = 0; v < 8; ++v) {
        float hv = c[v] > 0.0f ? c[v] : 0.0f;                       // ReLU
        hbuf[(mbase + v) * HID + nb * 16 + col] = (_Float16)hv;     // C layout -> row major
      }
    }
    asm volatile("s_wait_dscnt 0" ::: "memory");  // cross-lane RAW through LDS (same wave)

    // ================= Re-read h in A layout (transpose via LDS) =================
    v16h a2[2];
    #pragma unroll
    for (int kc = 0; kc < 2; ++kc) {
      v8h p0 = *(const v8h*)(hbuf + col * HID + kc * 32 + mbase);       // K kc*32+mbase+0..7
      v8h p1 = *(const v8h*)(hbuf + col * HID + kc * 32 + mbase + 16);  // K kc*32+mbase+16..23
      v16h t;
      #pragma unroll
      for (int i = 0; i < 8; ++i) { t[i] = p0[i]; t[8 + i] = p1[i]; }
      a2[kc] = t;
    }

    // dst node for each of this lane's 8 output rows
    int drow[8];
    #pragma unroll
    for (int v = 0; v < 8; ++v) {
      int er = tile * 16 + mbase + v;
      drow[v] = eidx[NE + (er < NE ? er : 0)];
    }

    // ================= GEMM2 (64->64) + bias, scatter atomic-max =================
    #pragma unroll
    for (int nb = 0; nb < 4; ++nb) {
      v8f c;
      #pragma unroll
      for (int v = 0; v < 8; ++v) c[v] = b2v[nb];
      c = __builtin_amdgcn_wmma_f32_16x16x32_f16(false, a2[0], false, bw2[0][nb],
                                                 (short)0, c, false, false);
      c = __builtin_amdgcn_wmma_f32_16x16x32_f16(false, a2[1], false, bw2[1][nb],
                                                 (short)0, c, false, false);
      #pragma unroll
      for (int v = 0; v < 8; ++v) {
        int er = tile * 16 + mbase + v;
        if (er < NE)
          atomic_max_f32(out + drow[v] * FOUT + nb * 16 + col, c[v]);
      }
    }
  }
}

extern "C" void kernel_launch(void* const* d_in, const int* in_sizes, int n_in,
                              void* d_out, int out_size, void* d_ws, size_t ws_size,
                              hipStream_t stream) {
  (void)in_sizes; (void)n_in; (void)d_ws; (void)ws_size;
  const float* x    = (const float*)d_in[0];
  const float* pos  = (const float*)d_in[1];
  const int*   eidx = (const int*)  d_in[2];
  const float* W1   = (const float*)d_in[3];
  const float* b1   = (const float*)d_in[4];
  const float* W2   = (const float*)d_in[5];
  const float* b2   = (const float*)d_in[6];
  float* out = (float*)d_out;

  const int n_out = NN * FOUT;  // == out_size
  (void)out_size;

  init_out_kernel<<<2048, 256, 0, stream>>>(out, n_out);
  pointconv_wmma_kernel<<<2048, 256, 0, stream>>>(x, pos, eidx, W1, b1, W2, b2, out);
  finalize_kernel<<<2048, 256, 0, stream>>>(out, n_out);
}
